// MvitVideoEncoder_38422777430163
// MI455X (gfx1250) — matrix operand order repaired and checked by reading, not compiled, not run on hardware
//
#include <hip/hip_runtime.h>
#include <hip/hip_bf16.h>
#include <math.h>

// ---------------------------------------------------------------------------
// Types
// ---------------------------------------------------------------------------
typedef _Float16 half_t;
typedef __attribute__((ext_vector_type(16))) _Float16 v16h;
typedef __attribute__((ext_vector_type(8)))  float    v8f;
typedef __attribute__((ext_vector_type(4)))  float    f32x4;

union F16Frag { v16h v; f32x4 f[2]; };

#define BATCH 2

// ---------------------------------------------------------------------------
// gfx1250 async global->LDS copy helpers (ASYNCcnt-tracked DMA, bypasses VGPRs)
// ---------------------------------------------------------------------------
__device__ __forceinline__ unsigned lds_off(const void* p)
{
    // generic -> LDS address space; ptrtoint of an AS(3) pointer is the
    // 32-bit LDS byte address used by async-to-LDS instructions.
    return (unsigned)(unsigned long long)(__attribute__((address_space(3))) const void*)p;
}

__device__ __forceinline__ void async_copy16(unsigned lds_addr, const void* gptr)
{
    unsigned long long ga = (unsigned long long)gptr;
    asm volatile("global_load_async_to_lds_b128 %0, %1, off"
                 :: "v"(lds_addr), "v"(ga) : "memory");
}

__device__ __forceinline__ void wait_async0()
{
    asm volatile("s_wait_asynccnt 0x0" ::: "memory");
}

// ---------------------------------------------------------------------------
// Generic WMMA GEMM:  C[b] = act( A[b] (MxK) * W[b]^T (NxK) * alpha + bias ) + R[b]
// A, W are f16 row-major (W is the weight matrix stored [N][K], i.e. B^T).
// Tiling: 128(M) x 64(N) x 32(K), 256 threads = 8 wave32 waves (4x2 grid of
// 32x32 wave tiles), double-buffered LDS staged with async-to-LDS DMA,
// v_wmma_f32_16x16x32_f16 with f32 accumulation.
// ---------------------------------------------------------------------------
#define TM 128
#define TN 64
#define TK 32
#define LDS_K 40   // 32 + 8 halves pad -> 80B row stride (16B aligned)

__global__ void __launch_bounds__(256)
wmma_gemm_kernel(const half_t* __restrict__ A, const half_t* __restrict__ Wt,
                 const float* __restrict__ bias, const float* __restrict__ R,
                 float* __restrict__ C, half_t* __restrict__ Ch,
                 int M, int N, int K,
                 int lda, int ldw, int ldr, int ldc, int ldch,
                 long long sA, long long sW, long long sR, long long sC, long long sCh,
                 float alpha, int act_gelu, int skipRow0)
{
    __shared__ half_t As[2][TM][LDS_K];
    __shared__ half_t Bs[2][TN][LDS_K];

    const int bz = blockIdx.z;
    A  += (long long)bz * sA;
    Wt += (long long)bz * sW;
    if (R)  R  += (long long)bz * sR;
    if (C)  C  += (long long)bz * sC;
    if (Ch) Ch += (long long)bz * sCh;

    const int m0 = blockIdx.y * TM;
    const int n0 = blockIdx.x * TN;
    const int tid = threadIdx.x;

    auto stage = [&](int s, int k0) {
        const bool kFull = (k0 + TK <= K);
        // ---- A tile: 128 rows x 32 halves, 16B chunks (512 chunks / 256 thr)
        if (kFull && m0 + TM <= M) {
            for (int c = tid; c < (TM * TK / 8); c += 256) {
                int row = c >> 2, kc = (c & 3) << 3;
                async_copy16(lds_off(&As[s][row][kc]),
                             A + (long long)(m0 + row) * lda + (k0 + kc));
            }
        } else {
            for (int c = tid; c < (TM * TK / 8); c += 256) {
                int row = c >> 2, kc = (c & 3) << 3;
                int gm = m0 + row, gk = k0 + kc;
                f32x4 val;
                if (gm < M && gk + 8 <= K) {
                    val = *reinterpret_cast<const f32x4*>(A + (long long)gm * lda + gk);
                } else {
                    alignas(16) half_t tmp[8];
                    for (int e = 0; e < 8; ++e)
                        tmp[e] = (gm < M && gk + e < K) ? A[(long long)gm * lda + gk + e]
                                                        : (half_t)0.0f;
                    val = *reinterpret_cast<const f32x4*>(tmp);
                }
                *reinterpret_cast<f32x4*>(&As[s][row][kc]) = val;
            }
        }
        // ---- W tile: 64 rows x 32 halves (256 chunks / 256 thr)
        if (kFull && n0 + TN <= N) {
            for (int c = tid; c < (TN * TK / 8); c += 256) {
                int row = c >> 2, kc = (c & 3) << 3;
                async_copy16(lds_off(&Bs[s][row][kc]),
                             Wt + (long long)(n0 + row) * ldw + (k0 + kc));
            }
        } else {
            for (int c = tid; c < (TN * TK / 8); c += 256) {
                int row = c >> 2, kc = (c & 3) << 3;
                int gn = n0 + row, gk = k0 + kc;
                f32x4 val;
                if (gn < N && gk + 8 <= K) {
                    val = *reinterpret_cast<const f32x4*>(Wt + (long long)gn * ldw + gk);
                } else {
                    alignas(16) half_t tmp[8];
                    for (int e = 0; e < 8; ++e)
                        tmp[e] = (gn < N && gk + e < K) ? Wt[(long long)gn * ldw + gk + e]
                                                        : (half_t)0.0f;
                    val = *reinterpret_cast<const f32x4*>(tmp);
                }
                *reinterpret_cast<f32x4*>(&Bs[s][row][kc]) = val;
            }
        }
    };

    const int lane = tid & 31, wave = tid >> 5;
    const int wm = (wave >> 1) * 32;   // wave M offset within tile (0..96)
    const int wn = (wave & 1) * 32;    // wave N offset within tile (0/32)
    const int hsel = lane >> 4;        // half-wave select
    const int l = lane & 15;

    v8f acc[2][2];
    for (int i = 0; i < 2; ++i)
        for (int j = 0; j < 2; ++j)
            for (int e = 0; e < 8; ++e) acc[i][j][e] = 0.0f;

    stage(0, 0);
    wait_async0();          // drain this wave's async-to-LDS DMA
    __syncthreads();

    const int nk = (K + TK - 1) / TK;
    for (int kt = 0; kt < nk; ++kt) {
        const int cur = kt & 1;
        if (kt + 1 < nk) stage(cur ^ 1, (kt + 1) * TK);

        F16Frag afrag[2], bfrag[2];
        // A frag (ISA 16-bit A layout): lane l = row M; K runs [hsel*8,+8) and [16+hsel*8,+8)
        for (int mi = 0; mi < 2; ++mi) {
            const half_t* ap = &As[cur][wm + mi * 16 + l][hsel * 8];
            afrag[mi].f[0] = *reinterpret_cast<const f32x4*>(ap);
            afrag[mi].f[1] = *reinterpret_cast<const f32x4*>(ap + 16);
        }
        // B frag (ISA 16-bit B layout): lane l = col N; K runs [hsel*8,+8) and [16+hsel*8,+8)
        for (int ni = 0; ni < 2; ++ni) {
            const half_t* bp = &Bs[cur][wn + ni * 16 + l][hsel * 8];
            bfrag[ni].f[0] = *reinterpret_cast<const f32x4*>(bp);
            bfrag[ni].f[1] = *reinterpret_cast<const f32x4*>(bp + 16);
        }
        for (int mi = 0; mi < 2; ++mi)
            for (int ni = 0; ni < 2; ++ni)
                acc[mi][ni] = __builtin_amdgcn_wmma_f32_16x16x32_f16(
                    false, afrag[mi].v, false, bfrag[ni].v,
                    (short)0, acc[mi][ni], false, false);

        wait_async0();      // next-buffer DMA must land before it is published
        __syncthreads();
    }

    // Epilogue. C/D layout: VGPR r, lanes[0:16) -> row hsel*8+r, col = lane&15
    for (int mi = 0; mi < 2; ++mi)
        for (int ni = 0; ni < 2; ++ni)
            for (int r = 0; r < 8; ++r) {
                int row = m0 + wm + mi * 16 + hsel * 8 + r;
                int col = n0 + wn + ni * 16 + l;
                if (row < M && col < N) {
                    float v = acc[mi][ni][r] * alpha;
                    if (bias) v += bias[col];
                    if (act_gelu) v = 0.5f * v * (1.0f + erff(v * 0.70710678118f));
                    if (R && !(skipRow0 && row == 0))
                        v += R[(long long)row * ldr + col];
                    if (C)  C[(long long)row * ldc + col] = v;
                    if (Ch) Ch[(long long)row * ldch + col] = (half_t)v;
                }
            }
}

// ---------------------------------------------------------------------------
// f32 -> f16 conversion (weights / shadows)
// ---------------------------------------------------------------------------
__global__ void f32_to_f16_kernel(const float* __restrict__ x, half_t* __restrict__ y,
                                  long long n)
{
    long long i = (long long)blockIdx.x * 256 + threadIdx.x;
    if (i < n) y[i] = (half_t)x[i];
}

// ---------------------------------------------------------------------------
// LayerNorm over last dim C (one block per row, 256 threads / 8 waves)
// ---------------------------------------------------------------------------
__global__ void __launch_bounds__(256)
layernorm_kernel(const float* __restrict__ x, const float* __restrict__ g,
                 const float* __restrict__ b,
                 float* __restrict__ y32, half_t* __restrict__ y16, int C)
{
    const long long row = blockIdx.x;
    const float* xr = x + row * C;
    __shared__ float red[8];
    __shared__ float stat[2];
    const int tid = threadIdx.x, lane = tid & 31, wv = tid >> 5;

    float s = 0.0f;
    for (int c = tid; c < C; c += 256) s += xr[c];
    for (int m = 16; m; m >>= 1) s += __shfl_xor(s, m, 32);
    if (!lane) red[wv] = s;
    __syncthreads();
    if (tid == 0) {
        float t = 0;
        for (int i = 0; i < 8; ++i) t += red[i];
        stat[0] = t / C;
    }
    __syncthreads();
    const float mean = stat[0];
    __syncthreads();

    float v = 0.0f;
    for (int c = tid; c < C; c += 256) { float d = xr[c] - mean; v += d * d; }
    for (int m = 16; m; m >>= 1) v += __shfl_xor(v, m, 32);
    if (!lane) red[wv] = v;
    __syncthreads();
    if (tid == 0) {
        float t = 0;
        for (int i = 0; i < 8; ++i) t += red[i];
        stat[1] = rsqrtf(t / C + 1e-6f);
    }
    __syncthreads();
    const float rs = stat[1];

    for (int c = tid; c < C; c += 256) {
        float o = (xr[c] - mean) * rs * g[c] + b[c];
        if (y32) y32[row * C + c] = o;
        if (y16) y16[row * C + c] = (half_t)o;
    }
}

// ---------------------------------------------------------------------------
// Depthwise 3x3x3 pooling conv (stride 1,s,s; pad 1) over one of q/k/v from
// the packed qkv buffer, followed by LayerNorm over head_dim.
// grid = (Nout, B*nH), block = hd threads.
// qkv layout: [(b*Nin + n) * 3*oc + sel*oc + h*hd + d]
// ---------------------------------------------------------------------------
__global__ void dw_pool_ln_kernel(const float* __restrict__ qkv,
                                  const float* __restrict__ pw,
                                  const float* __restrict__ g,
                                  const float* __restrict__ bln,
                                  float* __restrict__ out32,
                                  half_t* __restrict__ out16,
                                  half_t* __restrict__ out16t,
                                  int sel, int nH, int hd, int oc3,
                                  int Nin, int Nout,
                                  int T, int H, int W, int str,
                                  int oh2, int ow2, int ldt, float scale16)
{
    const int bh = blockIdx.y;
    const int bb = bh / nH, h = bh % nH;
    const int d = threadIdx.x;
    const int pos = blockIdx.x;
    const int chan = sel * (nH * hd) + h * hd + d;

    float val;
    if (pos == 0) {
        val = qkv[(long long)(bb * Nin) * oc3 + chan];
    } else {
        const int sp = pos - 1;
        const int ow = sp % ow2;
        const int oh = (sp / ow2) % oh2;
        const int ot = sp / (ow2 * oh2);
        const float* w = pw + d * 27;
        float acc = 0.0f;
        for (int kt = 0; kt < 3; ++kt) {
            int t = ot + kt - 1;                 // t-stride is always 1
            if (t < 0 || t >= T) continue;
            for (int ky = 0; ky < 3; ++ky) {
                int y = oh * str + ky - 1;
                if (y < 0 || y >= H) continue;
                for (int kx = 0; kx < 3; ++kx) {
                    int x = ow * str + kx - 1;
                    if (x < 0 || x >= W) continue;
                    int n = 1 + (t * H + y) * W + x;
                    acc += qkv[(long long)(bb * Nin + n) * oc3 + chan] *
                           w[(kt * 3 + ky) * 3 + kx];
                }
            }
        }
        val = acc;
    }

    // LayerNorm over hd (hd is a multiple of 32)
    float s1 = val, s2 = val * val;
    for (int m = 16; m; m >>= 1) {
        s1 += __shfl_xor(s1, m, 32);
        s2 += __shfl_xor(s2, m, 32);
    }
    __shared__ float w1[4], w2[4];
    const int lane = d & 31, wv = d >> 5, nw = blockDim.x >> 5;
    if (!lane) { w1[wv] = s1; w2[wv] = s2; }
    __syncthreads();
    float t1 = 0, t2 = 0;
    for (int i = 0; i < nw; ++i) { t1 += w1[i]; t2 += w2[i]; }
    const float mean = t1 / hd;
    const float var = t2 / hd - mean * mean;
    const float rs = rsqrtf(var + 1e-6f);
    const float out = (val - mean) * rs * g[d] + bln[d];

    const long long ob = ((long long)bh * Nout + pos) * hd + d;
    if (out32) out32[ob] = out;
    if (out16) out16[ob] = (half_t)(out * scale16);
    if (out16t) out16t[((long long)bh * hd + d) * ldt + pos] = (half_t)out;
}

// ---------------------------------------------------------------------------
// Decomposed relative position bias: attn[1+q, 1+k] += q . (Rt + Rh + Rw)
// grid = (Nq-1, B*nH), block = 128
// ---------------------------------------------------------------------------
__device__ __forceinline__ int rel_idx(int a, int qd, int kd, int b)
{
    float qr = (kd > qd) ? (float)kd / (float)qd : 1.0f;
    float kr = (qd > kd) ? (float)qd / (float)kd : 1.0f;
    return (int)(a * qr - b * kr + (kd - 1) * kr + 0.5f);
}

__global__ void __launch_bounds__(128)
relpos_kernel(const float* __restrict__ q32, float* __restrict__ attn,
              const float* __restrict__ relh, const float* __restrict__ relw,
              const float* __restrict__ relt,
              int Nq, int Nkv, int hd,
              int qt, int qh, int qw, int kt, int kh, int kw)
{
    __shared__ float qv[96];
    __shared__ float dt[128];
    __shared__ float dh[8];
    __shared__ float dw[8];

    const int bh = blockIdx.y;
    const int qpos = blockIdx.x;            // spatial (excl. cls)
    const int tid = threadIdx.x;

    const int iqw = qpos % qw;
    const int iqh = (qpos / qw) % qh;
    const int iqt = qpos / (qw * qh);

    if (tid < hd)
        qv[tid] = q32[((long long)bh * Nq + 1 + qpos) * hd + tid];
    __syncthreads();

    for (int i = tid; i < kt; i += blockDim.x) {
        const float* r = relt + (long long)rel_idx(iqt, qt, kt, i) * hd;
        float s = 0;
        for (int d = 0; d < hd; ++d) s += qv[d] * r[d];
        dt[i] = s;
    }
    for (int i = tid; i < kh; i += blockDim.x) {
        const float* r = relh + (long long)rel_idx(iqh, qh, kh, i) * hd;
        float s = 0;
        for (int d = 0; d < hd; ++d) s += qv[d] * r[d];
        dh[i] = s;
    }
    for (int i = tid; i < kw; i += blockDim.x) {
        const float* r = relw + (long long)rel_idx(iqw, qw, kw, i) * hd;
        float s = 0;
        for (int d = 0; d < hd; ++d) s += qv[d] * r[d];
        dw[i] = s;
    }
    __syncthreads();

    float* arow = attn + ((long long)bh * Nq + 1 + qpos) * Nkv + 1;
    const int nk = kt * kh * kw;
    for (int kp = tid; kp < nk; kp += blockDim.x) {
        const int ikw = kp % kw;
        const int ikh = (kp / kw) % kh;
        const int ikt = kp / (kw * kh);
        arow[kp] += dt[ikt] + dh[ikh] + dw[ikw];
    }
}

// ---------------------------------------------------------------------------
// Row softmax f32 -> f16 (padded row stride ldh). grid = (Nq, B*nH)
// ---------------------------------------------------------------------------
__global__ void __launch_bounds__(256)
softmax_kernel(const float* __restrict__ attn, half_t* __restrict__ out,
               int Nq, int Nkv, int ldh)
{
    const int bh = blockIdx.y, r = blockIdx.x;
    const float* row = attn + ((long long)bh * Nq + r) * Nkv;
    half_t* orow = out + ((long long)bh * Nq + r) * ldh;
    __shared__ float red[8];
    __shared__ float stat[2];
    const int tid = threadIdx.x, lane = tid & 31, wv = tid >> 5;

    float mx = -3.0e38f;
    for (int c = tid; c < Nkv; c += 256) mx = fmaxf(mx, row[c]);
    for (int m = 16; m; m >>= 1) mx = fmaxf(mx, __shfl_xor(mx, m, 32));
    if (!lane) red[wv] = mx;
    __syncthreads();
    if (tid == 0) {
        float t = -3.0e38f;
        for (int i = 0; i < 8; ++i) t = fmaxf(t, red[i]);
        stat[0] = t;
    }
    __syncthreads();
    const float gmx = stat[0];
    __syncthreads();

    float s = 0;
    for (int c = tid; c < Nkv; c += 256) s += expf(row[c] - gmx);
    for (int m = 16; m; m >>= 1) s += __shfl_xor(s, m, 32);
    if (!lane) red[wv] = s;
    __syncthreads();
    if (tid == 0) {
        float t = 0;
        for (int i = 0; i < 8; ++i) t += red[i];
        stat[1] = 1.0f / t;
    }
    __syncthreads();
    const float inv = stat[1];
    for (int c = tid; c < Nkv; c += 256)
        orow[c] = (half_t)(expf(row[c] - gmx) * inv);
}

// ---------------------------------------------------------------------------
// Max-pool skip connection: (B,1+T*H*W,C) -> (B,1+T*h2*w2,C), 3x3 window,
// stride 2, pad 1 in H/W; T passthrough; cls passthrough.  grid=(Nout,B)
// ---------------------------------------------------------------------------
__global__ void __launch_bounds__(256)
maxpool_skip_kernel(const float* __restrict__ x, float* __restrict__ y,
                    int C, int T, int H, int W, int h2, int w2)
{
    const int b = blockIdx.y;
    const int pos = blockIdx.x;
    const int Nin = 1 + T * H * W;
    const int Nout = 1 + T * h2 * w2;
    const float* xb = x + (long long)b * Nin * C;
    float* yb = y + (long long)b * Nout * C;

    if (pos == 0) {
        for (int c = threadIdx.x; c < C; c += 256) yb[c] = xb[c];
        return;
    }
    const int sp = pos - 1;
    const int ow = sp % w2;
    const int oh = (sp / w2) % h2;
    const int t = sp / (w2 * h2);
    for (int c = threadIdx.x; c < C; c += 256) {
        float m = -3.0e38f;
        for (int ky = 0; ky < 3; ++ky) {
            int hy = oh * 2 + ky - 1;
            if (hy < 0 || hy >= H) continue;
            for (int kx = 0; kx < 3; ++kx) {
                int wx = ow * 2 + kx - 1;
                if (wx < 0 || wx >= W) continue;
                int n = 1 + (t * H + hy) * W + wx;
                m = fmaxf(m, xb[(long long)n * C + c]);
            }
        }
        yb[(long long)pos * C + c] = m;
    }
}

// ---------------------------------------------------------------------------
// (B,nH,Nq,hd) f32 -> (B,Nq,nH*hd) f16 (attention output -> token major)
// grid = (Nq, B)
// ---------------------------------------------------------------------------
__global__ void __launch_bounds__(256)
o_transpose_kernel(const float* __restrict__ o32, half_t* __restrict__ ot,
                   int nH, int hd, int Nq)
{
    const int b = blockIdx.y, n = blockIdx.x;
    const int oc = nH * hd;
    for (int c = threadIdx.x; c < oc; c += 256) {
        const int h = c / hd, d = c % hd;
        ot[((long long)b * Nq + n) * oc + c] =
            (half_t)o32[(((long long)(b * nH + h)) * Nq + n) * hd + d];
    }
}

// ---------------------------------------------------------------------------
// Conv stem: video (B,3,128,96,96) * conv_w (96,3,3,15,15), stride (1,12,12),
// pad (1,3,3) -> tokens x0 (B, 8193, 96).  grid = (128*8*8, B), block = 128
// ---------------------------------------------------------------------------
__global__ void __launch_bounds__(128)
conv_stem_kernel(const float* __restrict__ video, const float* __restrict__ w,
                 const float* __restrict__ bias, float* __restrict__ x0)
{
    __shared__ float patch[2025];   // 3ic * 3t * 15 * 15
    const int bpos = blockIdx.x;
    const int b = blockIdx.y;
    const int ox = bpos % 8;
    const int oy = (bpos / 8) % 8;
    const int t = bpos / 64;
    const int tid = threadIdx.x;

    for (int i = tid; i < 2025; i += 128) {
        const int kx = i % 15;
        const int ky = (i / 15) % 15;
        const int kt = (i / 225) % 3;
        const int ic = i / 675;
        const int tt = t + kt - 1;
        const int yy = oy * 12 + ky - 3;
        const int xx = ox * 12 + kx - 3;
        float v = 0.0f;
        if (tt >= 0 && tt < 128 && yy >= 0 && yy < 96 && xx >= 0 && xx < 96)
            v = video[(((long long)(b * 3 + ic)) * 128 + tt) * 9216 + yy * 96 + xx];
        patch[i] = v;
    }
    __syncthreads();

    if (tid < 96) {
        const float* wr = w + (long long)tid * 2025;
        float acc = bias[tid];
        for (int i = 0; i < 2025; ++i) acc += patch[i] * wr[i];
        x0[((long long)b * 8193 + 1 + bpos) * 96 + tid] = acc;
    }
}

__global__ void cls_init_kernel(const float* __restrict__ cls, float* __restrict__ x0)
{
    const int i = blockIdx.x * 256 + threadIdx.x;
    if (i < BATCH * 96)
        x0[(long long)(i / 96) * 8193 * 96 + (i % 96)] = cls[i % 96];
}

// Final: ln (B,129,256) -> out (B,256,128) = x[:,1:].transpose(0,2,1)
__global__ void __launch_bounds__(256)
final_store_kernel(const float* __restrict__ ln, float* __restrict__ out)
{
    const int n = blockIdx.x, b = blockIdx.y;
    for (int c = threadIdx.x; c < 256; c += 256)
        out[((long long)(b * 256 + c)) * 128 + n] =
            ln[((long long)(b * 129 + 1 + n)) * 256 + c];
}

// ---------------------------------------------------------------------------
// Host orchestration
// ---------------------------------------------------------------------------
struct BlkCfg { int nH, ic, oc, hd, sq, skv, res; };
static const BlkCfg CFG[10] = {
    {1,  96,  96, 96, 1, 8, 0},
    {2,  96, 192, 96, 2, 4, 1},
    {2, 192, 192, 96, 1, 4, 0},
    {4, 192, 384, 96, 2, 2, 1},
    {4, 384, 384, 96, 1, 2, 0},
    {4, 384, 384, 96, 1, 2, 0},
    {4, 384, 384, 96, 1, 2, 0},
    {4, 384, 384, 96, 1, 2, 0},
    {8, 384, 768, 96, 2, 1, 1},
    {8, 768, 256, 32, 1, 1, 1},
};

// sorted per-block param indices (jax pytree dict-key order)
enum {
    P_MLP_B1 = 0, P_MLP_B2, P_MLP_W1, P_MLP_W2,
    P_N1_B, P_N1_G, P_N2_B, P_N2_G,
    P_PK_B, P_PK_G, P_PK_W,
    P_PQ_B, P_PQ_G, P_PQ_W,
    P_PROJ_B, P_PROJ_W,
    P_PV_B, P_PV_G, P_PV_W,
    P_QKV_B, P_QKV_W,
    P_REL_H, P_REL_T, P_REL_W,
    P_RES_B, P_RES_W
};

static inline int cdiv(int a, int b) { return (a + b - 1) / b; }
static inline int rnd8(int x) { return (x + 7) & ~7; }

static void gemm(hipStream_t st,
                 const half_t* A, const half_t* W, const float* bias, const float* R,
                 float* C, half_t* Ch, int M, int N, int K,
                 int lda, int ldw, int ldr, int ldc, int ldch,
                 long long sA, long long sW, long long sR, long long sC, long long sCh,
                 int batch, float alpha, int gelu_f, int skip0)
{
    dim3 g(cdiv(N, TN), cdiv(M, TM), batch);
    wmma_gemm_kernel<<<g, 256, 0, st>>>(A, W, bias, R, C, Ch, M, N, K,
                                        lda, ldw, ldr, ldc, ldch,
                                        sA, sW, sR, sC, sCh,
                                        alpha, gelu_f, skip0);
}

static void cvt16(hipStream_t st, const float* x, half_t* y, long long n)
{
    f32_to_f16_kernel<<<dim3((unsigned)((n + 255) / 256)), 256, 0, st>>>(x, y, n);
}

extern "C" void kernel_launch(void* const* d_in, const int* in_sizes, int n_in,
                              void* d_out, int out_size, void* d_ws, size_t ws_size,
                              hipStream_t stream)
{
    (void)in_sizes; (void)n_in; (void)out_size; (void)ws_size;

    // ---- map inputs (pytree sorted-key flatten order) ----
    const float* video = (const float*)d_in[0];
    const float* P[10][26];
    int pi = 1;
    for (int i = 0; i < 10; ++i) {
        const int np = CFG[i].res ? 26 : 24;
        for (int j = 0; j < np; ++j) P[i][j] = (const float*)d_in[pi++];
    }
    const float* cls    = (const float*)d_in[pi++];
    const float* conv_b = (const float*)d_in[pi++];
    const float* conv_w = (const float*)d_in[pi++];
    const float* norm_b = (const float*)d_in[pi++];
    const float* norm_g = (const float*)d_in[pi++];

    // ---- workspace bump allocator ----
    char* wbase = (char*)d_ws;
    size_t woff = 0;
    auto walloc = [&](size_t bytes) -> void* {
        woff = (woff + 255) & ~(size_t)255;
        void* p = wbase + woff;
        woff += bytes;
        return p;
    };

    // persistent ping-pong token buffers (max B * 8193 * 96 floats)
    const size_t slot = (size_t)BATCH * 8193 * 96 * sizeof(float);
    float* X[2] = { (float*)walloc(slot), (float*)walloc(slot) };
    const size_t mark = woff;

    // ---- conv stem ----
    conv_stem_kernel<<<dim3(8192, BATCH), 128, 0, stream>>>(video, conv_w, conv_b, X[0]);
    cls_init_kernel<<<1, 256, 0, stream>>>(cls, X[0]);

    float* x = X[0];
    int T = 128, H = 8;

    for (int i = 0; i < 10; ++i) {
        woff = mark;
        const BlkCfg& c = CFG[i];
        const int nH = c.nH, ic = c.ic, oc = c.oc, hd = c.hd;
        const int qH = (H - 1) / c.sq + 1;
        const int kH = (H - 1) / c.skv + 1;
        const int Nin = 1 + T * H * H;
        const int Nq  = 1 + T * qH * qH;
        const int Nkv = 1 + T * kH * kH;
        const int oc3 = 3 * oc;
        const int ldkv = rnd8(Nkv);
        const int BH = BATCH * nH;
        const float qscale = 1.0f / sqrtf((float)hd);

        // allocations
        half_t* xn1    = (half_t*)walloc((size_t)BATCH * Nin * ic * 2);
        half_t* qkvw16 = (half_t*)walloc((size_t)oc3 * ic * 2);
        float*  qkv32  = (float*) walloc((size_t)BATCH * Nin * oc3 * 4);
        float*  q32    = (float*) walloc((size_t)BH * Nq * hd * 4);
        half_t* q16    = (half_t*)walloc((size_t)BH * Nq * hd * 2);
        half_t* k16    = (half_t*)walloc((size_t)BH * Nkv * hd * 2);
        half_t* v16t   = (half_t*)walloc((size_t)BH * hd * ldkv * 2);
        float*  attn32 = (float*) walloc((size_t)BH * Nq * Nkv * 4);
        half_t* attn16 = (half_t*)walloc((size_t)BH * Nq * ldkv * 2);
        float*  o32    = (float*) walloc((size_t)BH * Nq * hd * 4);
        half_t* otok16 = (half_t*)walloc((size_t)BATCH * Nq * oc * 2);
        half_t* projw16= (half_t*)walloc((size_t)oc * oc * 2);
        float*  xblk   = (float*) walloc((size_t)BATCH * Nq * oc * 4);
        half_t* xn2    = (half_t*)walloc((size_t)BATCH * Nq * oc * 2);
        half_t* w116   = (half_t*)walloc((size_t)4 * oc * oc * 2);
        half_t* h16    = (half_t*)walloc((size_t)BATCH * Nq * 4 * oc * 2);
        half_t* w216   = (half_t*)walloc((size_t)oc * 4 * oc * 2);

        // 1. LN1 -> f16
        layernorm_kernel<<<BATCH * Nin, 256, 0, stream>>>(
            x, P[i][P_N1_G], P[i][P_N1_B], nullptr, xn1, ic);

        // 2. qkv GEMM
        cvt16(stream, P[i][P_QKV_W], qkvw16, (long long)oc3 * ic);
        gemm(stream, xn1, qkvw16, P[i][P_QKV_B], nullptr, qkv32, nullptr,
             BATCH * Nin, oc3, ic, ic, ic, 0, oc3, 0,
             0, 0, 0, 0, 0, 1, 1.0f, 0, 0);

        // 3. pooled q/k/v (+ per-head LN)
        dw_pool_ln_kernel<<<dim3(Nq, BH), hd, 0, stream>>>(
            qkv32, P[i][P_PQ_W], P[i][P_PQ_G], P[i][P_PQ_B],
            q32, q16, nullptr, 0, nH, hd, oc3, Nin, Nq,
            T, H, H, c.sq, qH, qH, 0, qscale);
        dw_pool_ln_kernel<<<dim3(Nkv, BH), hd, 0, stream>>>(
            qkv32, P[i][P_PK_W], P[i][P_PK_G], P[i][P_PK_B],
            nullptr, k16, nullptr, 1, nH, hd, oc3, Nin, Nkv,
            T, H, H, c.skv, kH, kH, 0, 1.0f);
        dw_pool_ln_kernel<<<dim3(Nkv, BH), hd, 0, stream>>>(
            qkv32, P[i][P_PV_W], P[i][P_PV_G], P[i][P_PV_B],
            nullptr, nullptr, v16t, 2, nH, hd, oc3, Nin, Nkv,
            T, H, H, c.skv, kH, kH, ldkv, 1.0f);

        // 4. attention scores (batched over B*nH); q pre-scaled
        gemm(stream, q16, k16, nullptr, nullptr, attn32, nullptr,
             Nq, Nkv, hd, hd, hd, 0, Nkv, 0,
             (long long)Nq * hd, (long long)Nkv * hd, 0, (long long)Nq * Nkv, 0,
             BH, 1.0f, 0, 0);

        // 5. relative position bias
        relpos_kernel<<<dim3(Nq - 1, BH), 128, 0, stream>>>(
            q32, attn32, P[i][P_REL_H], P[i][P_REL_W], P[i][P_REL_T],
            Nq, Nkv, hd, T, qH, qH, T, kH, kH);

        // 6. softmax -> f16
        softmax_kernel<<<dim3(Nq, BH), 256, 0, stream>>>(attn32, attn16, Nq, Nkv, ldkv);

        // 7. attn @ v + q residual (skip cls row)
        gemm(stream, attn16, v16t, nullptr, q32, o32, nullptr,
             Nq, hd, Nkv, ldkv, ldkv, hd, hd, 0,
             (long long)Nq * ldkv, (long long)hd * ldkv,
             (long long)Nq * hd, (long long)Nq * hd, 0,
             BH, 1.0f, 0, 1);

        // 8. (B,nH,Nq,hd) -> (B,Nq,oc) f16
        o_transpose_kernel<<<dim3(Nq, BATCH), 256, 0, stream>>>(o32, otok16, nH, hd, Nq);

        // 9. skip path: optional res projection + optional spatial max-pool
        float* src = x;
        if (c.res) {
            half_t* resw16 = (half_t*)walloc((size_t)oc * ic * 2);
            float* xres = (float*)walloc((size_t)BATCH * Nin * oc * 4);
            cvt16(stream, P[i][P_RES_W], resw16, (long long)oc * ic);
            gemm(stream, xn1, resw16, P[i][P_RES_B], nullptr, xres, nullptr,
                 BATCH * Nin, oc, ic, ic, ic, 0, oc, 0,
                 0, 0, 0, 0, 0, 1, 1.0f, 0, 0);
            src = xres;
        }
        float* xskip = src;
        if (c.sq > 1) {
            xskip = (float*)walloc((size_t)BATCH * Nq * oc * 4);
            maxpool_skip_kernel<<<dim3(Nq, BATCH), 256, 0, stream>>>(
                src, xskip, oc, T, H, H, qH, qH);
        }

        // 10. proj GEMM (+ skip residual)
        cvt16(stream, P[i][P_PROJ_W], projw16, (long long)oc * oc);
        gemm(stream, otok16, projw16, P[i][P_PROJ_B], xskip, xblk, nullptr,
             BATCH * Nq, oc, oc, oc, oc, oc, oc, 0,
             0, 0, 0, 0, 0, 1, 1.0f, 0, 0);

        // 11. LN2 -> f16
        layernorm_kernel<<<BATCH * Nq, 256, 0, stream>>>(
            xblk, P[i][P_N2_G], P[i][P_N2_B], nullptr, xn2, oc);

        // 12. MLP up + GELU (f16 out only)
        cvt16(stream, P[i][P_MLP_W1], w116, (long long)4 * oc * oc);
        gemm(stream, xn2, w116, P[i][P_MLP_B1], nullptr, nullptr, h16,
             BATCH * Nq, 4 * oc, oc, oc, oc, 0, 0, 4 * oc,
             0, 0, 0, 0, 0, 1, 1.0f, 1, 0);

        // 13. MLP down + residual -> next x (f32)
        cvt16(stream, P[i][P_MLP_W2], w216, (long long)oc * 4 * oc);
        float* xout = X[(i + 1) & 1];
        gemm(stream, h16, w216, P[i][P_MLP_B2], xblk, xout, nullptr,
             BATCH * Nq, oc, 4 * oc, 4 * oc, 4 * oc, oc, oc, 0,
             0, 0, 0, 0, 0, 1, 1.0f, 0, 0);

        x = xout;
        H = qH;
    }

    // ---- final LN + transpose store ----
    woff = mark;
    float* lnf = (float*)walloc((size_t)BATCH * 129 * 256 * 4);
    layernorm_kernel<<<BATCH * 129, 256, 0, stream>>>(x, norm_g, norm_b, lnf, nullptr, 256);
    final_store_kernel<<<dim3(128, BATCH), 256, 0, stream>>>(lnf, (float*)d_out);
}
